// MSAColumnGlobalAttention_16054587752603
// MI455X (gfx1250) — matrix operand order repaired and checked by reading, not compiled, not run on hardware
//
#include <hip/hip_runtime.h>

// ---------------------------------------------------------------------------
// MSAColumnGlobalAttention fused kernel for gfx1250 (MI455X, wave32, WMMA)
// One workgroup per residue r. LN(M) kept as f16 in the 320KB WGP LDS.
// Gate (M@Wg) and output (go@Wo) projections run on v_wmma_f32_16x16x32_f16.
// ---------------------------------------------------------------------------

typedef _Float16 v16h __attribute__((ext_vector_type(16)));
typedef float    v8f  __attribute__((ext_vector_type(8)));
typedef _Float16 h2   __attribute__((ext_vector_type(2)));

namespace {
constexpr int S  = 2048;
constexpr int R  = 384;
constexpr int D  = 64;
constexpr int NH = 8;
constexpr int C  = 8;
constexpr int HC = 64;

constexpr int NT = 256;   // threads per block
constexpr int NW = 8;     // waves per block (wave32)

constexpr int MLN_STRIDE = 68;  // halves/row: 64 data + 1 mask + pad (8B-aligned rows)
constexpr int STG_STRIDE = 72;  // go staging row stride (halves)
constexpr int W_STRIDE   = 80;  // transposed weight row stride (halves, 16B aligned)

constexpr size_t OFF_MLN = 0;
constexpr size_t OFF_STG = OFF_MLN + (size_t)S * MLN_STRIDE * 2;        // 278528
constexpr size_t OFF_WGT = OFF_STG + (size_t)NW * 16 * STG_STRIDE * 2;  // +18432
constexpr size_t OFF_WOT = OFF_WGT + (size_t)64 * W_STRIDE * 2;         // +10240
constexpr size_t OFF_KV  = OFF_WOT + (size_t)64 * W_STRIDE * 2;         // +10240
constexpr size_t OFF_F32 = OFF_KV + (size_t)64 * 16 * 4;                // +4096
// f32 region: lng[64] lnb[64] bg[64] bo[64] qsum[64] q[64] o[64] merge[8*80] cnt[1](+pad)
constexpr size_t SMEM_BYTES = OFF_F32 + (size_t)(7 * 64 + NW * 80 + 4) * 4;  // 325904 <= 320KB
}

__device__ __forceinline__ v8f wmma_f16(v16h a, v16h b, v8f acc) {
  return __builtin_amdgcn_wmma_f32_16x16x32_f16(false, a, false, b,
                                                (short)0, acc, false, false);
}

// Fast sigmoid: v_exp_f32 + v_rcp_f32 (avoid the IEEE div expansion).
__device__ __forceinline__ float fast_sigmoid(float z) {
  return __builtin_amdgcn_rcpf(1.0f + __expf(-z));
}

// A fragment (16x32 f16): lane L holds row (L&15); halves[0..15] = K0..K0+15,
// with K0 = 32*chunk + 16*(L>>4).
__device__ __forceinline__ v16h load_afrag(const _Float16* rowp, int K0) {
  union { v16h v; uint2 q[4]; } u;
  const uint2* p0 = reinterpret_cast<const uint2*>(rowp + K0);
  const uint2* p1 = reinterpret_cast<const uint2*>(rowp + K0 + 8);
  u.q[0] = p0[0]; u.q[1] = p0[1]; u.q[2] = p1[0]; u.q[3] = p1[1];
  return u.v;
}

// B fragment (32x16 f16): lane L holds column (L&15); halves[0..15] =
// K = kbase .. kbase+15 with kbase = 32*chunk + 16*(L>>4). Weights are stored
// transposed in LDS (row = N, 16 consecutive K per lane -> contiguous halves).
__device__ __forceinline__ v16h load_bfrag(const _Float16* wt, int n, int kbase) {
  union { v16h v; uint2 q[4]; } u;
  const uint2* p = reinterpret_cast<const uint2*>(wt + n * W_STRIDE + kbase);
  u.q[0] = p[0]; u.q[1] = p[1]; u.q[2] = p[2]; u.q[3] = p[3];
  return u.v;
}

__global__ __launch_bounds__(NT, 1) void msa_col_attn_kernel(
    const float* __restrict__ Mraw,   // [S,R,D]
    const float* __restrict__ Mmask,  // [S,R]
    const float* __restrict__ lng,    // [D]
    const float* __restrict__ lnb,    // [D]
    const float* __restrict__ Wq,     // [D,HC]
    const float* __restrict__ Wk,     // [D,C]
    const float* __restrict__ Wv,     // [D,C]
    const float* __restrict__ Wg,     // [D,HC]
    const float* __restrict__ bg,     // [HC]
    const float* __restrict__ Wo,     // [HC,D]
    const float* __restrict__ bo,     // [D]
    float* __restrict__ out)          // [S,R,D]
{
  extern __shared__ char smem[];
  _Float16* sMln = reinterpret_cast<_Float16*>(smem + OFF_MLN);
  _Float16* sStg = reinterpret_cast<_Float16*>(smem + OFF_STG);
  _Float16* sWgT = reinterpret_cast<_Float16*>(smem + OFF_WGT);
  _Float16* sWoT = reinterpret_cast<_Float16*>(smem + OFF_WOT);
  float*    sKV  = reinterpret_cast<float*>(smem + OFF_KV);   // [64][16]: k cols 0-7, v cols 8-15
  float*    sF   = reinterpret_cast<float*>(smem + OFF_F32);
  float* sLng  = sF;            // 64
  float* sLnb  = sLng + 64;     // 64
  float* sBg   = sLnb + 64;     // 64
  float* sBo   = sBg + 64;      // 64
  float* sQsum = sBo + 64;      // 64
  float* sQ    = sQsum + 64;    // 64
  float* sO    = sQ + 64;       // 64
  float* sMrg  = sO + 64;       // 8*80
  float* sCnt  = sMrg + NW*80;  // 1

  const int tid  = threadIdx.x;
  const int lane = tid & 31;
  const int wv   = tid >> 5;
  const int r    = blockIdx.x;

  // ---- preamble: stage weights/biases into LDS -----------------------------
  if (tid < 64) {
    sLng[tid] = lng[tid]; sLnb[tid] = lnb[tid];
    sBg[tid]  = bg[tid];  sBo[tid]  = bo[tid];
    sQsum[tid] = 0.0f;
  }
  if (tid == 0) sCnt[0] = 0.0f;
  for (int idx = tid; idx < 64 * 16; idx += NT) {
    int d = idx >> 4, c = idx & 15;
    sKV[idx] = (c < 8) ? Wk[d * C + c] : Wv[d * C + (c - 8)];
  }
  for (int idx = tid; idx < D * HC; idx += NT) {
    int d = idx >> 6, hc = idx & 63;
    sWgT[hc * W_STRIDE + d] = (_Float16)Wg[idx];          // Wg[d][hc] -> WgT[hc][d]
  }
  for (int idx = tid; idx < HC * D; idx += NT) {
    int hc = idx >> 6, d = idx & 63;
    sWoT[d * W_STRIDE + hc] = (_Float16)Wo[idx];          // Wo[hc][d] -> WoT[d][hc]
  }
  __syncthreads();

  // ---- pass A: LayerNorm rows -> f16 LDS tile; masked query pooling --------
  float qacc[64];
#pragma unroll
  for (int d = 0; d < 64; ++d) qacc[d] = 0.0f;
  float cntacc = 0.0f;

  for (int it = 0; it < S / NT; ++it) {
    const int s = it * NT + tid;
    // prefetch next row this thread will LayerNorm (global_prefetch_b8)
    if (it + 1 < S / NT)
      __builtin_prefetch(Mraw + ((size_t)(s + NT) * R + r) * D, 0, 0);

    float x[64];
    const float4* rp = reinterpret_cast<const float4*>(Mraw + ((size_t)s * R + r) * D);
#pragma unroll
    for (int j = 0; j < 16; ++j) {
      float4 t = rp[j];
      x[4*j] = t.x; x[4*j+1] = t.y; x[4*j+2] = t.z; x[4*j+3] = t.w;
    }
    float sum = 0.0f, sq = 0.0f;
#pragma unroll
    for (int d = 0; d < 64; ++d) { sum += x[d]; sq += x[d] * x[d]; }
    const float mu   = sum * (1.0f / 64.0f);
    const float var  = sq * (1.0f / 64.0f) - mu * mu;
    const float rstd = rsqrtf(var + 1e-5f);
    const float mval = Mmask[(size_t)s * R + r];

    _Float16* mrow = sMln + (size_t)s * MLN_STRIDE;
#pragma unroll
    for (int d = 0; d < 64; d += 2) {
      float a0 = (x[d]   - mu) * rstd * sLng[d]   + sLnb[d];
      float a1 = (x[d+1] - mu) * rstd * sLng[d+1] + sLnb[d+1];
      h2 p; p.x = (_Float16)a0; p.y = (_Float16)a1;
      *reinterpret_cast<h2*>(mrow + d) = p;
      qacc[d]   += a0 * mval;
      qacc[d+1] += a1 * mval;
    }
    mrow[64] = (_Float16)mval;   // stash mask in the row pad
    cntacc += mval;
  }
  // wave butterfly + one LDS atomic per wave per dim
#pragma unroll
  for (int d = 0; d < 64; ++d) {
    float v = qacc[d];
    for (int off = 16; off; off >>= 1) v += __shfl_xor(v, off, 32);
    if (lane == 0) atomicAdd(&sQsum[d], v);
  }
  {
    float cv = cntacc;
    for (int off = 16; off; off >>= 1) cv += __shfl_xor(cv, off, 32);
    if (lane == 0) atomicAdd(sCnt, cv);
  }
  __syncthreads();

  // ---- pass B: q projection + flash-style global attention over S ----------
  if (tid < HC) {
    const float inv = __builtin_amdgcn_rcpf(sCnt[0] + 1e-10f);
    float acc = 0.0f;
#pragma unroll 8
    for (int d = 0; d < 64; ++d) acc += sQsum[d] * inv * Wq[d * HC + tid];
    sQ[tid] = acc * 0.35355339059327373f;  // * C^-0.5
  }
  __syncthreads();

  float oacc[NH][C];
  float mh[NH], lh[NH];
#pragma unroll
  for (int h = 0; h < NH; ++h) {
    mh[h] = -1e30f; lh[h] = 0.0f;
#pragma unroll
    for (int c = 0; c < C; ++c) oacc[h][c] = 0.0f;
  }

  for (int s = tid; s < S; s += NT) {
    const _Float16* mrow = sMln + (size_t)s * MLN_STRIDE;
    float kk[C] = {0,0,0,0,0,0,0,0};
    float vv[C] = {0,0,0,0,0,0,0,0};
#pragma unroll 8
    for (int d = 0; d < 64; d += 2) {
      h2 p = *reinterpret_cast<const h2*>(mrow + d);
      float x0 = (float)p.x, x1 = (float)p.y;
      const float* w0 = sKV + d * 16;
      const float* w1 = sKV + (d + 1) * 16;
#pragma unroll
      for (int c = 0; c < C; ++c) {
        kk[c] += x0 * w0[c]     + x1 * w1[c];
        vv[c] += x0 * w0[8 + c] + x1 * w1[8 + c];
      }
    }
    const float mval = (float)mrow[64];
    const float bias = (mval - 1.0f) * 1e9f;
#pragma unroll
    for (int h = 0; h < NH; ++h) {
      float dot = 0.0f;
#pragma unroll
      for (int c = 0; c < C; ++c) dot += sQ[h * C + c] * kk[c];
      const float logit = dot + bias;
      const float mn = fmaxf(mh[h], logit);
      const float sc = __expf(mh[h] - mn);
      const float pe = __expf(logit - mn);
      lh[h] = lh[h] * sc + pe;
#pragma unroll
      for (int c = 0; c < C; ++c) oacc[h][c] = oacc[h][c] * sc + pe * vv[c];
      mh[h] = mn;
    }
  }
  // butterfly merge of softmax states within the wave
#pragma unroll
  for (int off = 1; off < 32; off <<= 1) {
#pragma unroll
    for (int h = 0; h < NH; ++h) {
      float mo = __shfl_xor(mh[h], off, 32);
      float lo = __shfl_xor(lh[h], off, 32);
      float mn = fmaxf(mh[h], mo);
      float sa = __expf(mh[h] - mn);
      float sb = __expf(mo - mn);
      lh[h] = lh[h] * sa + lo * sb;
#pragma unroll
      for (int c = 0; c < C; ++c) {
        float oo = __shfl_xor(oacc[h][c], off, 32);
        oacc[h][c] = oacc[h][c] * sa + oo * sb;
      }
      mh[h] = mn;
    }
  }
  if (lane == 0) {
    float* mw = sMrg + wv * 80;
#pragma unroll
    for (int h = 0; h < NH; ++h) { mw[h] = mh[h]; mw[8 + h] = lh[h]; }
#pragma unroll
    for (int h = 0; h < NH; ++h)
#pragma unroll
      for (int c = 0; c < C; ++c) mw[16 + h * C + c] = oacc[h][c];
  }
  __syncthreads();
  if (tid < HC) {
    const int h = tid >> 3;
    float M = -1e30f;
#pragma unroll
    for (int w = 0; w < NW; ++w) M = fmaxf(M, sMrg[w * 80 + h]);
    float L = 0.0f, O = 0.0f;
#pragma unroll
    for (int w = 0; w < NW; ++w) {
      const float e = __expf(sMrg[w * 80 + h] - M);
      L += sMrg[w * 80 + 8 + h] * e;
      O += sMrg[w * 80 + 16 + tid] * e;
    }
    sO[tid] = O * __builtin_amdgcn_rcpf(fmaxf(L, 1e-30f));
  }
  __syncthreads();

  // ---- pass C: WMMA gate projection, sigmoid gate, WMMA output projection --
  const int rrow = lane & 15;
  const int hi   = lane >> 4;
  const int kb   = hi * 16;
  _Float16* myStg = sStg + (size_t)wv * 16 * STG_STRIDE;

  for (int t = wv; t < S / 16; t += NW) {
    const int s0 = t * 16;
    const _Float16* arow = sMln + (size_t)(s0 + rrow) * MLN_STRIDE;
    const v16h a0 = load_afrag(arow, kb);        // K = 0..31
    const v16h a1 = load_afrag(arow, 32 + kb);   // K = 32..63

#pragma unroll
    for (int nt = 0; nt < 4; ++nt) {
      const int n = nt * 16 + rrow;
      v8f acc = {};
      acc = wmma_f16(a0, load_bfrag(sWgT, n, hi * 16), acc);
      acc = wmma_f16(a1, load_bfrag(sWgT, n, 32 + hi * 16), acc);
      const float og  = sO[n];
      const float bgv = sBg[n];
#pragma unroll
      for (int i = 0; i < 8; ++i) {
        const float g = fast_sigmoid(acc[i] + bgv);
        myStg[(i + 8 * hi) * STG_STRIDE + n] = (_Float16)(g * og);
      }
    }
    // same-wave LDS RAW: wait for the staging stores before fragment reads
    asm volatile("s_wait_dscnt 0" ::: "memory");
    __builtin_amdgcn_wave_barrier();

    const _Float16* grow = myStg + rrow * STG_STRIDE;
    const v16h g0 = load_afrag(grow, kb);
    const v16h g1 = load_afrag(grow, 32 + kb);

#pragma unroll
    for (int nt = 0; nt < 4; ++nt) {
      const int d = nt * 16 + rrow;
      v8f acc = {};
      acc = wmma_f16(g0, load_bfrag(sWoT, d, hi * 16), acc);
      acc = wmma_f16(g1, load_bfrag(sWoT, d, 32 + hi * 16), acc);
      const float bov = sBo[d];
#pragma unroll
      for (int i = 0; i < 8; ++i) {
        const int s = s0 + i + 8 * hi;
        const size_t gi = ((size_t)s * R + r) * D + d;
        out[gi] = Mraw[gi] + acc[i] + bov;
      }
    }
    asm volatile("" ::: "memory");  // keep next-iter staging stores after reads
  }
}

extern "C" void kernel_launch(void* const* d_in, const int* in_sizes, int n_in,
                              void* d_out, int out_size, void* d_ws, size_t ws_size,
                              hipStream_t stream) {
  (void)in_sizes; (void)n_in; (void)out_size; (void)d_ws; (void)ws_size;
  const float* Mraw  = (const float*)d_in[0];
  const float* Mmask = (const float*)d_in[1];
  const float* lng   = (const float*)d_in[2];
  const float* lnb   = (const float*)d_in[3];
  const float* Wq    = (const float*)d_in[4];
  const float* Wk    = (const float*)d_in[5];
  const float* Wv    = (const float*)d_in[6];
  const float* Wg    = (const float*)d_in[7];
  const float* bgp   = (const float*)d_in[8];
  const float* Wo    = (const float*)d_in[9];
  const float* bop   = (const float*)d_in[10];
  float* outp = (float*)d_out;

  (void)hipFuncSetAttribute(reinterpret_cast<const void*>(&msa_col_attn_kernel),
                            hipFuncAttributeMaxDynamicSharedMemorySize,
                            (int)SMEM_BYTES);
  hipLaunchKernelGGL(msa_col_attn_kernel, dim3(R), dim3(NT), SMEM_BYTES, stream,
                     Mraw, Mmask, lng, lnb, Wq, Wk, Wv, Wg, bgp, Wo, bop, outp);
}
